// Encoder_38809324487184
// MI455X (gfx1250) — compile-verified, matrix operand order checked by reading, not run on hardware
//
#include <hip/hip_runtime.h>

#define N_NODES    100000
#define N_EDGES    1600000
#define DIM        64
#define NLAYERS    5
#define NUM_GRAPHS 256
#define OUT_ROW    (DIM * NLAYERS)          // 320
#define POOL_CNT   (NUM_GRAPHS * OUT_ROW)   // 81920 floats, pooled block at start of d_out
#define BN_EPS     1e-5f

typedef __attribute__((ext_vector_type(16))) __bf16 v16bf;
typedef __attribute__((ext_vector_type(8)))  float  v8f;

__device__ __forceinline__ v8f wmma_bf16(v16bf a, v16bf b, v8f c) {
  // D = A(16x32 bf16) * B(32x16 bf16) + C(16x16 f32)
  return __builtin_amdgcn_wmma_f32_16x16x32_bf16(false, a, false, b, (short)0, c, false, false);
}

// Weights are staged into LDS pre-swizzled into the per-lane WMMA B-operand layout:
// packed[((s*4 + ntile)*32 + lane)*16 + i] = W[k][n]  with
//   s = k>>5, kk = k&31, g = (kk>>3)&1, i = (kk&7) + 8*(kk>>4),
//   lane = (n&15) + 16*g, ntile = n>>4.
// Then a B tile load is one aligned 32-byte vector read (2x ds_load_b128).
__device__ __forceinline__ v16bf load_btile(const __bf16* P, int t, int s, int lane) {
  return *(const v16bf*)(P + (((s * 4 + t) * 32 + lane) << 4));
}

__global__ void zero_kernel(float* __restrict__ p, int n) {
  for (int i = blockIdx.x * blockDim.x + threadIdx.x; i < n; i += gridDim.x * blockDim.x)
    p[i] = 0.0f;
}

__global__ void counts_kernel(const long long* __restrict__ batch, float* __restrict__ counts) {
  int i = blockIdx.x * blockDim.x + threadIdx.x;
  if (i < N_NODES) unsafeAtomicAdd(&counts[(int)batch[i]], 1.0f);
}

// One 16-lane group per edge; each lane handles 4 channels (float4 gather, 4 L2 atomics).
__global__ __launch_bounds__(256) void scatter_kernel(const float* __restrict__ h,
                                                      const long long* __restrict__ ei,
                                                      float* __restrict__ agg) {
  long long t = (long long)blockIdx.x * blockDim.x + threadIdx.x;
  long long e = t >> 4;
  if (e >= N_EDGES) return;
  int c4  = (int)(t & 15) * 4;
  int src = (int)ei[e];
  int dst = (int)ei[N_EDGES + e];
  float4 v = *(const float4*)(h + (long long)src * DIM + c4);
  float* p = agg + (long long)dst * DIM + c4;
  unsafeAtomicAdd(p + 0, v.x);
  unsafeAtomicAdd(p + 1, v.y);
  unsafeAtomicAdd(p + 2, v.z);
  unsafeAtomicAdd(p + 3, v.w);
}

// Fused: z = relu(relu((h+agg)W1+b1)W2+b2); writes z, accumulates per-channel sum/sumsq.
// 128 threads = 4 waves; each wave owns a 16-node M tile (block = 64 nodes).
__global__ __launch_bounds__(128) void mlp_kernel(const float* __restrict__ h,
                                                  const float* __restrict__ agg,
                                                  const float* __restrict__ W1,
                                                  const float* __restrict__ b1,
                                                  const float* __restrict__ W2,
                                                  const float* __restrict__ b2,
                                                  float* __restrict__ z,
                                                  float* __restrict__ gsum,
                                                  float* __restrict__ gsq) {
  __shared__ __attribute__((aligned(32))) __bf16 w1p[DIM * DIM];  // B-operand-swizzled
  __shared__ __attribute__((aligned(32))) __bf16 w2p[DIM * DIM];
  __shared__ float tbuf[4][16 * 68];   // per-wave D->A transpose tile, padded stride 68
  __shared__ float ssum[DIM], ssq[DIM];

  const int tid = threadIdx.x;
  // Stage weights to LDS as bf16, pre-swizzled into WMMA B-register order.
  for (int idx = tid; idx < DIM * DIM; idx += 128) {
    const int k = idx >> 6, n = idx & 63;
    const int s = k >> 5, kk = k & 31;
    const int gg = (kk >> 3) & 1;
    const int ii = (kk & 7) + 8 * (kk >> 4);
    const int dst = (((s * 4 + (n >> 4)) * 32 + ((n & 15) + 16 * gg)) << 4) + ii;
    w1p[dst] = (__bf16)W1[idx];
    w2p[dst] = (__bf16)W2[idx];
  }
  if (tid < DIM) { ssum[tid] = 0.0f; ssq[tid] = 0.0f; }
  __syncthreads();

  const int wave = tid >> 5, lane = tid & 31;
  const int m = lane & 15, g = lane >> 4;
  const int nb = blockIdx.x * 64 + wave * 16;
  float* tw = &tbuf[wave][0];

  if (nb < N_NODES) {  // N_NODES % 16 == 0 -> full M tiles only (wave-uniform predicate)
    const float* hp = h + (nb + m) * DIM;
    const float* ap = agg + (nb + m) * DIM;
    v16bf a0, a1;
    {
      const int o0 = 8 * g, o1 = 16 + 8 * g;
#pragma unroll
      for (int i = 0; i < 8; ++i) {
        a0[i]     = (__bf16)(hp[o0 + i]      + ap[o0 + i]);
        a0[i + 8] = (__bf16)(hp[o1 + i]      + ap[o1 + i]);
        a1[i]     = (__bf16)(hp[32 + o0 + i] + ap[32 + o0 + i]);
        a1[i + 8] = (__bf16)(hp[32 + o1 + i] + ap[32 + o1 + i]);
      }
    }
    // GEMM1: 4 N tiles x 2 K steps, bias+ReLU, spill to LDS in (row, col) order
#pragma unroll
    for (int t = 0; t < 4; ++t) {
      const int n = t * 16 + m;
      v8f c = {};
      c = wmma_bf16(a0, load_btile(w1p, t, 0, lane), c);
      c = wmma_bf16(a1, load_btile(w1p, t, 1, lane), c);
      const float bias = b1[n];
#pragma unroll
      for (int r = 0; r < 8; ++r) {            // D row M = 8*g + r, col = n
        float v = c[r] + bias;
        tw[(8 * g + r) * 68 + n] = v > 0.0f ? v : 0.0f;
      }
    }
    asm volatile("s_wait_dscnt 0" ::: "memory");  // cross-lane LDS transpose hand-off
    // Reload as A-operand (row m per lane)
    {
      const float* tr = &tw[m * 68];
      const int o0 = 8 * g, o1 = 16 + 8 * g;
#pragma unroll
      for (int i = 0; i < 8; ++i) {
        a0[i]     = (__bf16)tr[o0 + i];
        a0[i + 8] = (__bf16)tr[o1 + i];
        a1[i]     = (__bf16)tr[32 + o0 + i];
        a1[i + 8] = (__bf16)tr[32 + o1 + i];
      }
    }
    // GEMM2 + bias + ReLU + store + BN stats
#pragma unroll
    for (int t = 0; t < 4; ++t) {
      const int n = t * 16 + m;
      v8f c = {};
      c = wmma_bf16(a0, load_btile(w2p, t, 0, lane), c);
      c = wmma_bf16(a1, load_btile(w2p, t, 1, lane), c);
      const float bias = b2[n];
      float s = 0.0f, s2 = 0.0f;
#pragma unroll
      for (int r = 0; r < 8; ++r) {
        float v = fmaxf(c[r] + bias, 0.0f);
        z[(nb + 8 * g + r) * DIM + n] = v;
        s += v; s2 += v * v;
      }
      unsafeAtomicAdd(&ssum[n], s);   // ds_add_f32
      unsafeAtomicAdd(&ssq[n], s2);
    }
  }
  __syncthreads();
  if (tid < DIM)            unsafeAtomicAdd(&gsum[tid], ssum[tid]);
  else if (tid < 2 * DIM)   unsafeAtomicAdd(&gsq[tid - DIM], ssq[tid - DIM]);
}

__global__ void bn_finalize_kernel(const float* __restrict__ gsum, const float* __restrict__ gsq,
                                   const float* __restrict__ gamma, const float* __restrict__ beta,
                                   float* __restrict__ bnab, int layer) {
  int c = threadIdx.x;
  float mu  = gsum[c] * (1.0f / N_NODES);
  float var = fmaxf(gsq[c] * (1.0f / N_NODES) - mu * mu, 0.0f);
  float a   = gamma[layer * DIM + c] * rsqrtf(var + BN_EPS);
  bnab[c]       = a;
  bnab[DIM + c] = beta[layer * DIM + c] - mu * a;
}

// Normalize, write h_next + xs slab of d_out, and atomically mean-pool into d_out head.
__global__ __launch_bounds__(256) void bn_apply_kernel(const float* __restrict__ z,
                                                       const float* __restrict__ bnab,
                                                       const long long* __restrict__ batch,
                                                       float* __restrict__ hnext,
                                                       float* __restrict__ out, int layer) {
  int idx = blockIdx.x * blockDim.x + threadIdx.x;   // node*16 + quad
  int node = idx >> 4;
  if (node >= N_NODES) return;
  int c4 = (idx & 15) * 4;
  float4 zv = *(const float4*)(z + node * DIM + c4);
  float4 av = *(const float4*)(bnab + c4);
  float4 bv = *(const float4*)(bnab + DIM + c4);
  float4 r;
  r.x = zv.x * av.x + bv.x;
  r.y = zv.y * av.y + bv.y;
  r.z = zv.z * av.z + bv.z;
  r.w = zv.w * av.w + bv.w;
  *(float4*)(hnext + node * DIM + c4) = r;
  *(float4*)(out + POOL_CNT + node * OUT_ROW + layer * DIM + c4) = r;
  float* p = out + (int)batch[node] * OUT_ROW + layer * DIM + c4;
  unsafeAtomicAdd(p + 0, r.x);
  unsafeAtomicAdd(p + 1, r.y);
  unsafeAtomicAdd(p + 2, r.z);
  unsafeAtomicAdd(p + 3, r.w);
}

__global__ void pool_div_kernel(float* __restrict__ out, const float* __restrict__ counts) {
  int i = blockIdx.x * blockDim.x + threadIdx.x;
  if (i < POOL_CNT) out[i] /= fmaxf(counts[i / OUT_ROW], 1.0f);
}

extern "C" void kernel_launch(void* const* d_in, const int* in_sizes, int n_in,
                              void* d_out, int out_size, void* d_ws, size_t ws_size,
                              hipStream_t stream) {
  const float*     x     = (const float*)d_in[0];
  const long long* ei    = (const long long*)d_in[1];
  const long long* batch = (const long long*)d_in[2];
  const float*     W1    = (const float*)d_in[3];
  const float*     b1    = (const float*)d_in[4];
  const float*     W2    = (const float*)d_in[5];
  const float*     b2    = (const float*)d_in[6];
  const float*     gamma = (const float*)d_in[7];
  const float*     beta  = (const float*)d_in[8];
  float* out = (float*)d_out;

  float* ws     = (float*)d_ws;
  float* hbuf   = ws;                      // 6,400,000
  float* agg    = hbuf + N_NODES * DIM;    // 6,400,000
  float* zbuf   = agg + N_NODES * DIM;     // 6,400,000
  float* stats  = zbuf + N_NODES * DIM;    // gsum[64] + gsq[64]
  float* bnab   = stats + 2 * DIM;         // a[64] + b[64]
  float* counts = bnab + 2 * DIM;          // 256

  // Re-initialize all accumulators every call (graph replay safe).
  zero_kernel<<<512, 256, 0, stream>>>(out, POOL_CNT);
  zero_kernel<<<2, 256, 0, stream>>>(counts, NUM_GRAPHS);
  counts_kernel<<<(N_NODES + 255) / 256, 256, 0, stream>>>(batch, counts);

  for (int L = 0; L < NLAYERS; ++L) {
    const float* hcur = (L == 0) ? x : hbuf;
    zero_kernel<<<2048, 256, 0, stream>>>(agg, N_NODES * DIM);
    zero_kernel<<<1, 256, 0, stream>>>(stats, 2 * DIM);
    scatter_kernel<<<(N_EDGES * 16) / 256, 256, 0, stream>>>(hcur, ei, agg);
    mlp_kernel<<<(N_NODES + 63) / 64, 128, 0, stream>>>(
        hcur, agg, W1 + L * DIM * DIM, b1 + L * DIM, W2 + L * DIM * DIM, b2 + L * DIM,
        zbuf, stats, stats + DIM);
    bn_finalize_kernel<<<1, DIM, 0, stream>>>(stats, stats + DIM, gamma, beta, bnab, L);
    bn_apply_kernel<<<(N_NODES * 16 + 255) / 256, 256, 0, stream>>>(zbuf, bnab, batch, hbuf, out, L);
  }
  pool_div_kernel<<<(POOL_CNT + 255) / 256, 256, 0, stream>>>(out, counts);
}
